// HGT_iterative_79319456022564
// MI455X (gfx1250) — matrix operand order repaired and checked by reading, not compiled, not run on hardware
//
#include <hip/hip_runtime.h>
#include <hip/hip_bf16.h>
#include <math.h>

typedef __attribute__((ext_vector_type(16))) __bf16 v16bf;
typedef __attribute__((ext_vector_type(8)))  __bf16 v8bf;
typedef __attribute__((ext_vector_type(8)))  float  v8f;

#define AACT_NONE 0
#define AACT_GELU 1
#define EPI_NONE 0
#define EPI_RELU 1
#define EPI_SKIPMIX 2
#define BSRC_DENSE 0
#define BSRC_BLOCKDIAG 1

__device__ __forceinline__ float gelu_exact(float x) {
  return 0.5f * x * (1.0f + erff(x * 0.7071067811865475f));
}
// order-preserving monotonic encoding of float into uint (for atomic segment max)
__device__ __forceinline__ unsigned ford_enc(float x) {
  unsigned u = __float_as_uint(x);
  return (u & 0x80000000u) ? ~u : (u | 0x80000000u);
}
__device__ __forceinline__ float ford_dec(unsigned e) {
  unsigned u = (e & 0x80000000u) ? (e ^ 0x80000000u) : ~e;
  return __uint_as_float(u);
}
__device__ __forceinline__ void atomic_add_f32(float* p, float v) {
  __hip_atomic_fetch_add(p, v, __ATOMIC_RELAXED, __HIP_MEMORY_SCOPE_AGENT);
}

union PK8  { __bf16 h[8];  uint4 u; };
union PK16 { __bf16 h[16]; uint4 u[2]; };

// ---------------------------------------------------------------------------
// Fused GEMM: Y[M,Ncols] = EPI( AACT(X[M,128]) @ W[128,Ncols] + bias )
// bf16 WMMA 16x16x32, fp32 accumulate. One block = 128x128 output tile,
// 8 waves; each wave: 16 rows x 128 cols (8 n-subtiles of v8f accumulators).
// LDS tiles are in fragment-native layouts so operands come from wide,
// conflict-free 16B ds_load_b128 ops. Templated so each variant is
// straight-line code (no runtime-flag branches in the hot loop).
// BSRC==BLOCKDIAG: W is block-diagonal from arel[h][32][32] (per-head xform).
// ---------------------------------------------------------------------------
template <int AACT, int BSRC, int EPI>
__global__ __launch_bounds__(256) void gemm_bf16_wmma(
    const float* __restrict__ X, const float* __restrict__ Wd,
    const float* __restrict__ bias, const float* __restrict__ resid,
    const float* __restrict__ skipp, float* __restrict__ Y,
    int M, int Ncols)
{
  __shared__ __bf16 As[128][136];  // [m][K], row stride 272B (17*16B) ~34.8KB
  __shared__ __bf16 BsT[128][40];  // [n][k-in-chunk], row stride 80B   ~10.2KB

  const int t = threadIdx.x;
  const int mbase = blockIdx.x * 128;
  const int nbase = blockIdx.y * 128;

  // ---- stage A (full 128x128 tile), activation applied, packed 16B stores ----
  {
    const int r  = t >> 1;            // X row within tile
    const int c0 = (t & 1) * 64;      // K half
    const int grow = mbase + r;
    if (grow < M) {
      const float* xp = X + (size_t)grow * 128 + c0;
      #pragma unroll
      for (int j = 0; j < 64; j += 8) {
        float4 f0 = *(const float4*)(xp + j);
        float4 f1 = *(const float4*)(xp + j + 4);
        if (AACT == AACT_GELU) {
          f0.x = gelu_exact(f0.x); f0.y = gelu_exact(f0.y);
          f0.z = gelu_exact(f0.z); f0.w = gelu_exact(f0.w);
          f1.x = gelu_exact(f1.x); f1.y = gelu_exact(f1.y);
          f1.z = gelu_exact(f1.z); f1.w = gelu_exact(f1.w);
        }
        PK8 pk;
        pk.h[0] = (__bf16)f0.x; pk.h[1] = (__bf16)f0.y;
        pk.h[2] = (__bf16)f0.z; pk.h[3] = (__bf16)f0.w;
        pk.h[4] = (__bf16)f1.x; pk.h[5] = (__bf16)f1.y;
        pk.h[6] = (__bf16)f1.z; pk.h[7] = (__bf16)f1.w;
        *(uint4*)&As[r][c0 + j] = pk.u;
      }
    } else {
      const uint4 z = {0u, 0u, 0u, 0u};
      #pragma unroll
      for (int j = 0; j < 64; j += 8) *(uint4*)&As[r][c0 + j] = z;
    }
  }

  const int w     = t >> 5;       // wave id 0..7
  const int lane  = t & 31;
  const int lhalf = lane >> 4;    // 0 or 1
  const int l15   = lane & 15;
  const int m     = w * 16 + l15; // A row this lane contributes

  v8f acc[8] = {};

  for (int kc = 0; kc < 4; kc++) {
    __syncthreads();
    // ---- stage B chunk transposed: BsT[col][k] for k = kc*32..kc*32+31 ----
    {
      const int col   = t >> 1;         // 0..127 (output column within tile)
      const int khalf = (t & 1) * 16;   // 16 k values per thread
      PK16 pk;
      if (BSRC == BSRC_DENSE) {
        const int gc = nbase + col;
        if (gc < Ncols) {
          #pragma unroll
          for (int j = 0; j < 16; j++)
            pk.h[j] = (__bf16)Wd[(size_t)(kc * 32 + khalf + j) * Ncols + gc];
        } else {
          #pragma unroll
          for (int j = 0; j < 16; j++) pk.h[j] = (__bf16)0.0f;
        }
      } else {
        // block-diagonal: W[h*32+d][h*32+e] = arel[h][d][e].
        // Within this 32-aligned chunk, k>>5 == kc, so the diagonal test
        // is uniform across the whole 16-element run: one branch, not 16.
        const int hcol = col >> 5, ecol = col & 31;
        if (hcol == kc) {
          #pragma unroll
          for (int j = 0; j < 16; j++)
            pk.h[j] = (__bf16)Wd[hcol * 1024 + (khalf + j) * 32 + ecol];
        } else {
          #pragma unroll
          for (int j = 0; j < 16; j++) pk.h[j] = (__bf16)0.0f;
        }
      }
      *(uint4*)&BsT[col][khalf]     = pk.u[0];
      *(uint4*)&BsT[col][khalf + 8] = pk.u[1];
    }
    __syncthreads();

    // ---- A fragment: two contiguous 16B runs, concat in registers ----
    const int ab = kc * 32 + lhalf * 8;
    const v8bf alo = *(const v8bf*)&As[m][ab];
    const v8bf ahi = *(const v8bf*)&As[m][ab + 16];
    const v16bf a = __builtin_shufflevector(alo, ahi,
        0, 1, 2, 3, 4, 5, 6, 7, 8, 9, 10, 11, 12, 13, 14, 15);

    const int bb = lhalf * 8;  // K base within chunk for B fragment
    #pragma unroll
    for (int ns = 0; ns < 8; ns++) {
      const int n = ns * 16 + l15;
      const v8bf blo = *(const v8bf*)&BsT[n][bb];
      const v8bf bhi = *(const v8bf*)&BsT[n][bb + 16];
      const v16bf b = __builtin_shufflevector(blo, bhi,
          0, 1, 2, 3, 4, 5, 6, 7, 8, 9, 10, 11, 12, 13, 14, 15);
      acc[ns] = __builtin_amdgcn_wmma_f32_16x16x32_bf16(
          false, a, false, b, (short)0, acc[ns], false, false);
    }
  }

  // ---- epilogue ----
  float beta = 0.0f;
  if (EPI == EPI_SKIPMIX) beta = 1.0f / (1.0f + expf(-skipp[0]));
  #pragma unroll
  for (int ns = 0; ns < 8; ns++) {
    const int col = nbase + ns * 16 + l15;
    if (col >= Ncols) continue;
    const float bsv = bias ? bias[col] : 0.0f;
    #pragma unroll
    for (int r8 = 0; r8 < 8; r8++) {
      const int row = mbase + w * 16 + lhalf * 8 + r8;  // C/D layout: M = r + 8*lanehalf
      if (row >= M) continue;
      float v = acc[ns][r8] + bsv;
      if (EPI == EPI_RELU) v = fmaxf(v, 0.0f);
      else if (EPI == EPI_SKIPMIX)
        v = beta * v + (1.0f - beta) * resid[(size_t)row * 128 + col];
      Y[(size_t)row * Ncols + col] = v;
    }
  }
}

// ---------------------------------------------------------------------------
// fills
// ---------------------------------------------------------------------------
__global__ void fill_f32(float* __restrict__ p, float v, size_t n) {
  size_t i = (size_t)blockIdx.x * blockDim.x + threadIdx.x;
  const size_t st = (size_t)gridDim.x * blockDim.x;
  for (; i < n; i += st) p[i] = v;
}
__global__ void fill_u32(unsigned* __restrict__ p, unsigned v, size_t n) {
  size_t i = (size_t)blockIdx.x * blockDim.x + threadIdx.x;
  const size_t st = (size_t)gridDim.x * blockDim.x;
  for (; i < n; i += st) p[i] = v;
}

// ---------------------------------------------------------------------------
// edge kernels: one thread per (edge, head); 32-float rows per head
// ---------------------------------------------------------------------------
__global__ void edge_logits_kernel(
    const float* __restrict__ Qd, const float* __restrict__ KR,
    const int* __restrict__ src, const int* __restrict__ dst,
    const float* __restrict__ prel, unsigned* __restrict__ segmax,
    float* __restrict__ elog, int E)
{
  const int idx = blockIdx.x * blockDim.x + threadIdx.x;
  if (idx >= E * 4) return;
  const int e = idx >> 2, h = idx & 3;
  const int s = src[e], d = dst[e];
  const float4* q = (const float4*)(Qd + (size_t)d * 128 + h * 32);
  const float4* k = (const float4*)(KR + (size_t)s * 128 + h * 32);
  float acc = 0.0f;
  #pragma unroll
  for (int j = 0; j < 8; j++) {
    float4 a = q[j], b = k[j];
    acc += a.x * b.x + a.y * b.y + a.z * b.z + a.w * b.w;
  }
  const float logit = acc * prel[h] * 0.17677669529663687f;  // 1/sqrt(32)
  elog[idx] = logit;
  atomicMax(&segmax[d * 4 + h], ford_enc(logit));
}

__global__ void edge_exp_kernel(
    float* __restrict__ elog, const int* __restrict__ dst,
    const unsigned* __restrict__ segmax, float* __restrict__ segsum, int E)
{
  const int idx = blockIdx.x * blockDim.x + threadIdx.x;
  if (idx >= E * 4) return;
  const int e = idx >> 2, h = idx & 3;
  const int d = dst[e];
  const float mx = ford_dec(segmax[d * 4 + h]);
  const float ex = expf(elog[idx] - mx);
  elog[idx] = ex;
  atomic_add_f32(&segsum[d * 4 + h], ex);
}

__global__ void edge_scatter_kernel(
    const float* __restrict__ elog, const float* __restrict__ VR,
    const int* __restrict__ src, const int* __restrict__ dst,
    const float* __restrict__ segsum, float* __restrict__ OUTd, int E)
{
  const int idx = blockIdx.x * blockDim.x + threadIdx.x;
  if (idx >= E * 4) return;
  const int e = idx >> 2, h = idx & 3;
  const int s = src[e], d = dst[e];
  const float alpha = elog[idx] / segsum[d * 4 + h];
  const float4* v = (const float4*)(VR + (size_t)s * 128 + h * 32);
  float* o = OUTd + (size_t)d * 128 + h * 32;
  #pragma unroll
  for (int j = 0; j < 8; j++) {
    float4 x = v[j];
    atomic_add_f32(o + 4 * j + 0, x.x * alpha);
    atomic_add_f32(o + 4 * j + 1, x.y * alpha);
    atomic_add_f32(o + 4 * j + 2, x.z * alpha);
    atomic_add_f32(o + 4 * j + 3, x.w * alpha);
  }
}

// ---------------------------------------------------------------------------
extern "C" void kernel_launch(void* const* d_in, const int* in_sizes, int n_in,
                              void* d_out, int out_size, void* d_ws, size_t ws_size,
                              hipStream_t stream) {
  (void)in_sizes; (void)n_in; (void)out_size; (void)ws_size;

  const float* xfeat[4] = {(const float*)d_in[0], (const float*)d_in[1],
                           (const float*)d_in[2], (const float*)d_in[3]};
  const int* esrc[7] = {(const int*)d_in[4],  (const int*)d_in[6],  (const int*)d_in[8],
                        (const int*)d_in[10], (const int*)d_in[12], (const int*)d_in[14],
                        (const int*)d_in[16]};
  const int* edst[7] = {(const int*)d_in[5],  (const int*)d_in[7],  (const int*)d_in[9],
                        (const int*)d_in[11], (const int*)d_in[13], (const int*)d_in[15],
                        (const int*)d_in[17]};
  static const int ECNT[7] = {50000, 250000, 300000, 200000, 50000, 250000, 200000};
  static const int ES[7]   = {1, 1, 0, 0, 2, 0, 3};
  static const int ED[7]   = {2, 0, 0, 3, 1, 1, 0};
  static const int NN[4]   = {80000, 120000, 8000, 25000};
  static const long long OFF[4] = {0, 80000, 200000, 208000};
  const long long NTOT = 233000;

  const float* Wlin = (const float*)d_in[18];
  const float* blin = (const float*)d_in[19];
  const float* Wout = (const float*)d_in[44];
  const float* bout = (const float*)d_in[45];

  // workspace layout (floats)
  float* ws   = (float*)d_ws;
  float* xsA  = ws;
  float* xsB  = xsA  + NTOT * 128;
  float* Kb   = xsB  + NTOT * 128;
  float* Qb   = Kb   + NTOT * 128;
  float* Vb   = Qb   + NTOT * 128;
  float* OUTA = Vb   + NTOT * 128;
  float* KR   = OUTA + NTOT * 128;
  float* VR   = KR   + 120000LL * 128;
  float* ELOG = VR   + 120000LL * 128;
  float* SEGS = ELOG + 300000LL * 4;
  unsigned* SEGM = (unsigned*)(SEGS + 120000LL * 4);

  // templated GEMM dispatch (only 4 variants are used)
  auto GEMM = [&](const float* A, const float* Wp, const float* bp,
                  const float* res, const float* sk, float* Yp,
                  int M, int Nc, int aact, int bsrc, int epi) {
    dim3 g((unsigned)((M + 127) / 128), (unsigned)((Nc + 127) / 128), 1);
    dim3 b(256, 1, 1);
    if (aact == AACT_NONE && bsrc == BSRC_DENSE && epi == EPI_RELU)
      gemm_bf16_wmma<AACT_NONE, BSRC_DENSE, EPI_RELU>
          <<<g, b, 0, stream>>>(A, Wp, bp, res, sk, Yp, M, Nc);
    else if (aact == AACT_NONE && bsrc == BSRC_DENSE && epi == EPI_NONE)
      gemm_bf16_wmma<AACT_NONE, BSRC_DENSE, EPI_NONE>
          <<<g, b, 0, stream>>>(A, Wp, bp, res, sk, Yp, M, Nc);
    else if (aact == AACT_NONE && bsrc == BSRC_BLOCKDIAG && epi == EPI_NONE)
      gemm_bf16_wmma<AACT_NONE, BSRC_BLOCKDIAG, EPI_NONE>
          <<<g, b, 0, stream>>>(A, Wp, bp, res, sk, Yp, M, Nc);
    else
      gemm_bf16_wmma<AACT_GELU, BSRC_DENSE, EPI_SKIPMIX>
          <<<g, b, 0, stream>>>(A, Wp, bp, res, sk, Yp, M, Nc);
  };

  // ---- input projection: xsA[t] = relu(feat @ Wlin[t] + blin[t]) ----
  for (int t = 0; t < 4; t++)
    GEMM(xfeat[t], Wlin + (long long)t * 16384, blin + t * 128, nullptr, nullptr,
         xsA + OFF[t] * 128, NN[t], 128, AACT_NONE, BSRC_DENSE, EPI_RELU);

  float* xin = xsA;
  float* xout = xsB;
  for (int L = 0; L < 2; L++) {
    const int pb = 20 + 12 * L;
    const float* Wk   = (const float*)d_in[pb + 0];
    const float* bk   = (const float*)d_in[pb + 1];
    const float* Wq   = (const float*)d_in[pb + 2];
    const float* bq   = (const float*)d_in[pb + 3];
    const float* Wv   = (const float*)d_in[pb + 4];
    const float* bv   = (const float*)d_in[pb + 5];
    const float* Wa   = (const float*)d_in[pb + 6];
    const float* ba   = (const float*)d_in[pb + 7];
    const float* skip = (const float*)d_in[pb + 8];
    const float* arel = (const float*)d_in[pb + 9];
    const float* mrel = (const float*)d_in[pb + 10];
    const float* prel = (const float*)d_in[pb + 11];

    // K/Q/V projections per node type
    for (int t = 0; t < 4; t++) {
      GEMM(xin + OFF[t] * 128, Wk + (long long)t * 16384, bk + t * 128, nullptr, nullptr,
           Kb + OFF[t] * 128, NN[t], 128, AACT_NONE, BSRC_DENSE, EPI_NONE);
      GEMM(xin + OFF[t] * 128, Wq + (long long)t * 16384, bq + t * 128, nullptr, nullptr,
           Qb + OFF[t] * 128, NN[t], 128, AACT_NONE, BSRC_DENSE, EPI_NONE);
      GEMM(xin + OFF[t] * 128, Wv + (long long)t * 16384, bv + t * 128, nullptr, nullptr,
           Vb + OFF[t] * 128, NN[t], 128, AACT_NONE, BSRC_DENSE, EPI_NONE);
    }
    fill_f32<<<2048, 256, 0, stream>>>(OUTA, 0.0f, (size_t)(NTOT * 128));

    for (int r = 0; r < 7; r++) {
      const int s = ES[r], d = ED[r], E = ECNT[r];
      // per-relation per-head transforms as block-diagonal WMMA GEMMs
      GEMM(Kb + OFF[s] * 128, arel + (long long)r * 4096, nullptr, nullptr, nullptr,
           KR, NN[s], 128, AACT_NONE, BSRC_BLOCKDIAG, EPI_NONE);
      GEMM(Vb + OFF[s] * 128, mrel + (long long)r * 4096, nullptr, nullptr, nullptr,
           VR, NN[s], 128, AACT_NONE, BSRC_BLOCKDIAG, EPI_NONE);
      // segment softmax state: max = enc(-inf), sum = 0
      fill_u32<<<256, 256, 0, stream>>>(SEGM, 0x007FFFFFu, (size_t)(NN[d] * 4));
      fill_f32<<<256, 256, 0, stream>>>(SEGS, 0.0f, (size_t)(NN[d] * 4));
      const int nthr = E * 4;
      const int nb = (nthr + 255) / 256;
      edge_logits_kernel<<<nb, 256, 0, stream>>>(Qb + OFF[d] * 128, KR, esrc[r], edst[r],
                                                 prel + r * 4, SEGM, ELOG, E);
      edge_exp_kernel<<<nb, 256, 0, stream>>>(ELOG, edst[r], SEGM, SEGS, E);
      edge_scatter_kernel<<<nb, 256, 0, stream>>>(ELOG, VR, esrc[r], edst[r], SEGS,
                                                  OUTA + OFF[d] * 128, E);
    }

    // xout = sigmoid(skip)*(gelu(out)@Wa + ba) + (1-sigmoid(skip))*xin
    for (int t = 0; t < 4; t++)
      GEMM(OUTA + OFF[t] * 128, Wa + (long long)t * 16384, ba + t * 128,
           xin + OFF[t] * 128, skip + t, xout + OFF[t] * 128,
           NN[t], 128, AACT_GELU, BSRC_DENSE, EPI_SKIPMIX);

    float* tmp = xin; xin = xout; xout = tmp;
  }

  // final: paper features @ Wout + bout -> d_out [80000 x 349]
  GEMM(xin, Wout, bout, nullptr, nullptr, (float*)d_out,
       80000, 349, AACT_NONE, BSRC_DENSE, EPI_NONE);
}
